// SpatialSelfAttention_87694642250009
// MI455X (gfx1250) — matrix operand order repaired and checked by reading, not compiled, not run on hardware
//
#include <hip/hip_runtime.h>
#include <hip/hip_bf16.h>

typedef __bf16 bf16_t;
typedef __bf16 v16bf __attribute__((ext_vector_type(16)));
typedef float  v8f   __attribute__((ext_vector_type(8)));

namespace {
constexpr int kB    = 16;
constexpr int kC    = 512;
constexpr int kN    = 1024;   // H*W
constexpr int kCqk  = 64;     // C/8
constexpr int kMall = 640;    // 64 (q) + 64 (k) + 512 (v) stacked rows

constexpr int BM = 128;       // block tile M
constexpr int BN = 128;       // block tile N
constexpr int BK = 32;        // K step == WMMA K for bf16
constexpr int NT = 256;       // 8 waves of 32
}

// Flip to 0 if the assembler rejects the async mnemonics.
#define USE_ASYNC_COPY 1

// ---------------------------------------------------------------------------
// One 16-byte global->LDS transfer per lane.  Async path uses the gfx1250
// GLOBAL_LOAD_ASYNC_TO_LDS_B128 (ASYNCcnt-tracked, no VGPR round trip);
// fallback is a plain load+store pair.
// ---------------------------------------------------------------------------
__device__ __forceinline__ void copy_b128_to_lds(const bf16_t* g, bf16_t* l)
{
#if USE_ASYNC_COPY
  asm volatile("global_load_async_to_lds_b128 %0, %1, off"
               :: "v"((unsigned)(uintptr_t)l),
                  "v"((unsigned long long)(uintptr_t)g)
               : "memory");
#else
  *reinterpret_cast<uint4*>(l) = *reinterpret_cast<const uint4*>(g);
#endif
}

__device__ __forceinline__ void copy_wait()
{
#if USE_ASYNC_COPY
  asm volatile("s_wait_asynccnt 0x0" ::: "memory");
#endif
}

// ---------------------------------------------------------------------------
// WMMA bf16 mainloop.  Block tile BM(128) x BN(128), 8 waves in a 4(M) x 2(N)
// grid, each wave computing a 32x64 tile as 2x4 v_wmma_f32_16x16x32_bf16
// fragments.  BOTH operands arrive as row-major [row][k] 2D tiles (B is
// pre-transposed in global memory by the producer kernels), so LDS staging is
// pure 16B vector copies and double-buffered: stage t+1's async copies are in
// flight while stage t's WMMAs run.
//
// A fragment (16-bit 16x32, ISA 7.12.2): lane L (m = L&15, hi = L>>4),
//   element e -> k = e + (e>=8 ? 8 : 0) + hi*8   (two contiguous 8-elem runs)
// B fragment (16-bit 32x16): lane L (n = L&15, hi = L>>4),
//   element e -> k = e + (e>=8 ? 8 : 0) + hi*8   (two contiguous 8-elem runs)
// ---------------------------------------------------------------------------
__device__ __forceinline__ void gemm_mainloop_bf16(
    const bf16_t* __restrict__ A,  int lda,   // [M][K] row-major
    const bf16_t* __restrict__ Bt, int ldb,   // [N][K] row-major (B^T)
    int K, int block_m, int block_n,
    bf16_t* As, bf16_t* Bs,                   // 2*BM*BK and 2*BN*BK bf16
    v8f (&acc)[2][4])
{
  const int tid  = threadIdx.x;
  const int lane = tid & 31;
  const int wave = tid >> 5;
  const int wm   = wave & 3;   // 0..3  (M)
  const int wn   = wave >> 2;  // 0..1  (N)
  const int hi   = lane >> 4;
  const int l15  = lane & 15;

  auto load_stage = [&](int k0, int buf) {
    // A tile: BM x BK bf16 = 8KB -> 512 x 16B, 2 per thread
    #pragma unroll
    for (int c = 0; c < 2; ++c) {
      int idx = tid + c * NT;           // 0..511
      int m   = idx >> 2;               // 0..127
      int kc  = (idx & 3) * 8;          // 0,8,16,24
      copy_b128_to_lds(&A[(size_t)(block_m + m) * lda + k0 + kc],
                       &As[buf * (BM * BK) + m * BK + kc]);
    }
    // B^T tile: BN x BK bf16 = 8KB -> 512 x 16B, 2 per thread
    #pragma unroll
    for (int c = 0; c < 2; ++c) {
      int idx = tid + c * NT;
      int n   = idx >> 2;
      int kc  = (idx & 3) * 8;
      copy_b128_to_lds(&Bt[(size_t)(block_n + n) * ldb + k0 + kc],
                       &Bs[buf * (BN * BK) + n * BK + kc]);
    }
  };

  load_stage(0, 0);
  copy_wait();
  __syncthreads();

  const int nk = K / BK;
  for (int t = 0; t < nk; ++t) {
    const int buf = t & 1;
    if (t + 1 < nk) load_stage((t + 1) * BK, buf ^ 1);  // overlap with WMMAs

    const bf16_t* Ab = &As[buf * (BM * BK)];
    const bf16_t* Bb = &Bs[buf * (BN * BK)];

    v16bf afrag[2], bfrag[4];
    #pragma unroll
    for (int i = 0; i < 2; ++i) {
      int m = wm * 32 + i * 16 + l15;
      #pragma unroll
      for (int e = 0; e < 16; ++e) {
        int kk = e + ((e & 8) ? 8 : 0) + hi * 8;
        afrag[i][e] = Ab[m * BK + kk];
      }
    }
    #pragma unroll
    for (int j = 0; j < 4; ++j) {
      int n = wn * 64 + j * 16 + l15;
      #pragma unroll
      for (int e = 0; e < 16; ++e) {
        int kk = e + ((e & 8) ? 8 : 0) + hi * 8;
        bfrag[j][e] = Bb[n * BK + kk];
      }
    }
    #pragma unroll
    for (int i = 0; i < 2; ++i)
      #pragma unroll
      for (int j = 0; j < 4; ++j)
        acc[i][j] = __builtin_amdgcn_wmma_f32_16x16x32_bf16(
            false, afrag[i], false, bfrag[j], (short)0, acc[i][j], false, false);

    copy_wait();
    __syncthreads();
  }
}

// C/D fragment mapping (32-bit 16x16): lane L (n = L&15, hi = L>>4), vgpr r ->
// m = r + hi*8.
#define EPILOGUE_FOREACH(body)                                            \
  {                                                                       \
    const int lane = threadIdx.x & 31;                                    \
    const int wave = threadIdx.x >> 5;                                    \
    const int wm = wave & 3, wn = wave >> 2;                              \
    const int hi = lane >> 4, l15 = lane & 15;                            \
    _Pragma("unroll") for (int i = 0; i < 2; ++i)                         \
    _Pragma("unroll") for (int j = 0; j < 4; ++j)                         \
    _Pragma("unroll") for (int r = 0; r < 8; ++r) {                       \
      const int row = block_m + wm * 32 + i * 16 + r + hi * 8;            \
      const int col = block_n + wn * 64 + j * 16 + l15;                   \
      body                                                                \
    }                                                                     \
  }

#define DECLARE_ACC()                                                     \
  v8f acc[2][4];                                                          \
  for (int i = 0; i < 2; ++i)                                             \
    for (int j = 0; j < 4; ++j)                                           \
      for (int r = 0; r < 8; ++r) acc[i][j][r] = 0.0f;

// ---------------------------------------------------------------------------
// Pack kernels: weights (Wq;Wk;Wv stacked) -> bf16; x -> bf16 transposed to
// [b][n][c] so it is directly the [N][K] B^T operand of the projection GEMM.
// ---------------------------------------------------------------------------
__global__ __launch_bounds__(NT)
void pack_w_kernel(const float* __restrict__ Wq, const float* __restrict__ Wk,
                   const float* __restrict__ Wv, const float* __restrict__ bq,
                   const float* __restrict__ bk, const float* __restrict__ bv,
                   bf16_t* __restrict__ Wcat, float* __restrict__ bcat)
{
  int idx = blockIdx.x * blockDim.x + threadIdx.x;
  if (idx >= kMall * kC) return;
  int row = idx / kC, c = idx % kC;
  float v;
  if (row < kCqk)            v = Wq[row * kC + c];
  else if (row < 2 * kCqk)   v = Wk[(row - kCqk) * kC + c];
  else                       v = Wv[(row - 2 * kCqk) * kC + c];
  Wcat[idx] = (bf16_t)v;
  if (c == 0) {
    float bb;
    if (row < kCqk)          bb = bq[row];
    else if (row < 2 * kCqk) bb = bk[row - kCqk];
    else                     bb = bv[row - 2 * kCqk];
    bcat[row] = bb;
  }
}

__global__ __launch_bounds__(NT)
void pack_xT_kernel(const float* __restrict__ x, bf16_t* __restrict__ xbT)
{
  int idx = blockIdx.x * blockDim.x + threadIdx.x;   // over B*C*N, n contiguous
  if (idx >= kB * kC * kN) return;
  int b = idx / (kC * kN);
  int rem = idx % (kC * kN);
  int c = rem / kN;
  int n = rem % kN;
  xbT[((size_t)b * kN + n) * kC + c] = (bf16_t)x[idx];
}

__global__ __launch_bounds__(NT)
void zero_f32_kernel(float* __restrict__ p, int n)
{
  int idx = blockIdx.x * blockDim.x + threadIdx.x;
  if (idx < n) p[idx] = 0.0f;
}

// ---------------------------------------------------------------------------
// GEMM 1: [Wq;Wk;Wv] x X.  q and k both stored [n][d] (each directly an
// operand of the scores GEMM: q^T is A, k^T is B^T); v stored [c][n] (A of AV).
// ---------------------------------------------------------------------------
__global__ __launch_bounds__(NT)
void gemm_proj_kernel(const bf16_t* __restrict__ Wcat, const bf16_t* __restrict__ xbT,
                      const float* __restrict__ bcat,
                      bf16_t* __restrict__ qT, bf16_t* __restrict__ kT,
                      bf16_t* __restrict__ vM)
{
  __shared__ __attribute__((aligned(16))) bf16_t As[2 * BM * BK];
  __shared__ __attribute__((aligned(16))) bf16_t Bs[2 * BN * BK];
  const int b = blockIdx.z;
  const int block_m = blockIdx.y * BM;
  const int block_n = blockIdx.x * BN;
  DECLARE_ACC();

  gemm_mainloop_bf16(Wcat, kC, xbT + (size_t)b * kN * kC, kC, kC,
                     block_m, block_n, As, Bs, acc);

  EPILOGUE_FOREACH({
    float y = acc[i][j][r] + bcat[row];
    if (row < kCqk)
      qT[((size_t)b * kN + col) * kCqk + row] = (bf16_t)y;
    else if (row < 2 * kCqk)
      kT[((size_t)b * kN + col) * kCqk + (row - kCqk)] = (bf16_t)y;
    else
      vM[((size_t)b * kC + (row - 2 * kCqk)) * kN + col] = (bf16_t)y;
  })
}

// ---------------------------------------------------------------------------
// GEMM 2: S = (1/sqrt(C)) * q^T k, masked.  A = qT [N][64], B^T = kT [N][64].
// ---------------------------------------------------------------------------
__global__ __launch_bounds__(NT)
void gemm_scores_kernel(const bf16_t* __restrict__ qT, const bf16_t* __restrict__ kT,
                        const int* __restrict__ mask, float* __restrict__ scores)
{
  __shared__ __attribute__((aligned(16))) bf16_t As[2 * BM * BK];
  __shared__ __attribute__((aligned(16))) bf16_t Bs[2 * BN * BK];
  const int b = blockIdx.z;
  const int block_m = blockIdx.y * BM;
  const int block_n = blockIdx.x * BN;
  DECLARE_ACC();

  gemm_mainloop_bf16(qT + (size_t)b * kN * kCqk, kCqk,
                     kT + (size_t)b * kN * kCqk, kCqk, kCqk,
                     block_m, block_n, As, Bs, acc);

  const float scale = 0.044194173824159216f;  // 1/sqrt(512)
  EPILOGUE_FOREACH({
    size_t o = ((size_t)b * kN + row) * kN + col;
    float s = acc[i][j][r] * scale;
    if (mask[o] == 0) s = -1e20f;
    scores[o] = s;
  })
}

// ---------------------------------------------------------------------------
// Row softmax (twice) + vis column-sum accumulation.  attn is written in its
// natural [i][j] layout: exactly the [N][K] B^T operand of the AV GEMM, and
// the stores are contiguous.
// ---------------------------------------------------------------------------
__device__ __forceinline__ float block_red_max(float v, float* red)
{
  int tid = threadIdx.x;
  red[tid] = v; __syncthreads();
  for (int s = NT / 2; s > 0; s >>= 1) {
    if (tid < s) red[tid] = fmaxf(red[tid], red[tid + s]);
    __syncthreads();
  }
  float r = red[0]; __syncthreads();
  return r;
}

__device__ __forceinline__ float block_red_sum(float v, float* red)
{
  int tid = threadIdx.x;
  red[tid] = v; __syncthreads();
  for (int s = NT / 2; s > 0; s >>= 1) {
    if (tid < s) red[tid] = red[tid] + red[tid + s];
    __syncthreads();
  }
  float r = red[0]; __syncthreads();
  return r;
}

__global__ __launch_bounds__(NT)
void softmax_rows_kernel(const float* __restrict__ scores,
                         bf16_t* __restrict__ attnM, float* __restrict__ vis)
{
  constexpr int PT = kN / NT;  // 4 elements per thread
  const int i = blockIdx.x;
  const int b = blockIdx.y;
  const int tid = threadIdx.x;
  __shared__ float red[NT];

  const float* srow = scores + ((size_t)b * kN + i) * kN;
  float v[PT];
  float mx = -3.402823e38f;
  #pragma unroll
  for (int t = 0; t < PT; ++t) { v[t] = srow[tid + t * NT]; mx = fmaxf(mx, v[t]); }
  mx = block_red_max(mx, red);

  float sum = 0.0f;
  #pragma unroll
  for (int t = 0; t < PT; ++t) { v[t] = __expf(v[t] - mx); sum += v[t]; }
  sum = block_red_sum(sum, red);
  const float inv = 1.0f / sum;

  bf16_t* arow = attnM + ((size_t)b * kN + i) * kN;
  float a[PT];
  #pragma unroll
  for (int t = 0; t < PT; ++t) {
    a[t] = v[t] * inv;
    arow[tid + t * NT] = (bf16_t)a[t];          // contiguous row store
  }

  // second (redundant) softmax of the attn row -> attn2; vis[j] += attn2[i][j]
  float mx2 = -3.402823e38f;
  #pragma unroll
  for (int t = 0; t < PT; ++t) mx2 = fmaxf(mx2, a[t]);
  mx2 = block_red_max(mx2, red);
  float e2[PT], sum2 = 0.0f;
  #pragma unroll
  for (int t = 0; t < PT; ++t) { e2[t] = __expf(a[t] - mx2); sum2 += e2[t]; }
  sum2 = block_red_sum(sum2, red);
  const float inv2 = 1.0f / sum2;
  #pragma unroll
  for (int t = 0; t < PT; ++t)
    atomicAdd(&vis[b * kN + (tid + t * NT)], e2[t] * inv2);
}

// ---------------------------------------------------------------------------
// GEMM 3: out = gamma * (V x attn^T) + x.  A = vM [C][N], B^T = attn [N][N].
// ---------------------------------------------------------------------------
__global__ __launch_bounds__(NT)
void gemm_av_kernel(const bf16_t* __restrict__ vM, const bf16_t* __restrict__ attnM,
                    const float* __restrict__ x, const float* __restrict__ gamma,
                    float* __restrict__ out)
{
  __shared__ __attribute__((aligned(16))) bf16_t As[2 * BM * BK];
  __shared__ __attribute__((aligned(16))) bf16_t Bs[2 * BN * BK];
  const int b = blockIdx.z;
  const int block_m = blockIdx.y * BM;
  const int block_n = blockIdx.x * BN;
  DECLARE_ACC();

  gemm_mainloop_bf16(vM + (size_t)b * kC * kN, kN,
                     attnM + (size_t)b * kN * kN, kN, kN,
                     block_m, block_n, As, Bs, acc);

  const float g = gamma[0];
  EPILOGUE_FOREACH({
    size_t o = ((size_t)b * kC + row) * kN + col;
    out[o] = g * acc[i][j][r] + x[o];
  })
}

// ---------------------------------------------------------------------------
// vis normalization: vis[b, :] / max(vis[b, :])
// ---------------------------------------------------------------------------
__global__ __launch_bounds__(NT)
void vis_norm_kernel(const float* __restrict__ vis, float* __restrict__ visOut)
{
  constexpr int PT = kN / NT;
  const int b = blockIdx.x;
  const int tid = threadIdx.x;
  __shared__ float red[NT];
  float mx = -3.402823e38f;
  #pragma unroll
  for (int t = 0; t < PT; ++t) mx = fmaxf(mx, vis[b * kN + tid + t * NT]);
  mx = block_red_max(mx, red);
  const float inv = 1.0f / mx;
  #pragma unroll
  for (int t = 0; t < PT; ++t) {
    int j = tid + t * NT;
    visOut[b * kN + j] = vis[b * kN + j] * inv;
  }
}

// ---------------------------------------------------------------------------
extern "C" void kernel_launch(void* const* d_in, const int* in_sizes, int n_in,
                              void* d_out, int out_size, void* d_ws, size_t ws_size,
                              hipStream_t stream)
{
  const float* x     = (const float*)d_in[0];
  const int*   mask  = (const int*)d_in[1];
  const float* Wq    = (const float*)d_in[2];
  const float* bq    = (const float*)d_in[3];
  const float* Wk    = (const float*)d_in[4];
  const float* bk    = (const float*)d_in[5];
  const float* Wv    = (const float*)d_in[6];
  const float* bv    = (const float*)d_in[7];
  const float* gamma = (const float*)d_in[8];

  float* out    = (float*)d_out;                       // [B, C, H, W]
  float* visOut = out + (size_t)kB * kC * kN;          // [B, H, W]

  size_t off = 0;
  auto take = [&](size_t bytes) -> void* {
    void* p = (char*)d_ws + off;
    off += (bytes + 255) & ~(size_t)255;
    return p;
  };
  bf16_t* Wcat   = (bf16_t*)take((size_t)kMall * kC * sizeof(bf16_t));
  float*  bcat   = (float*) take((size_t)kMall * sizeof(float));
  bf16_t* xbT    = (bf16_t*)take((size_t)kB * kN * kC * sizeof(bf16_t));
  bf16_t* qT     = (bf16_t*)take((size_t)kB * kN * kCqk * sizeof(bf16_t));
  bf16_t* kT     = (bf16_t*)take((size_t)kB * kN * kCqk * sizeof(bf16_t));
  bf16_t* vM     = (bf16_t*)take((size_t)kB * kC * kN * sizeof(bf16_t));
  float*  scores = (float*) take((size_t)kB * kN * kN * sizeof(float));
  bf16_t* attnM  = (bf16_t*)take((size_t)kB * kN * kN * sizeof(bf16_t));
  float*  vis    = (float*) take((size_t)kB * kN * sizeof(float));
  (void)ws_size; (void)in_sizes; (void)n_in; (void)out_size;

  pack_w_kernel<<<(kMall * kC + NT - 1) / NT, NT, 0, stream>>>(
      Wq, Wk, Wv, bq, bk, bv, Wcat, bcat);
  pack_xT_kernel<<<(kB * kC * kN + NT - 1) / NT, NT, 0, stream>>>(x, xbT);
  zero_f32_kernel<<<(kB * kN + NT - 1) / NT, NT, 0, stream>>>(vis, kB * kN);

  gemm_proj_kernel<<<dim3(kN / BN, kMall / BM, kB), NT, 0, stream>>>(
      Wcat, xbT, bcat, qT, kT, vM);

  gemm_scores_kernel<<<dim3(kN / BN, kN / BM, kB), NT, 0, stream>>>(
      qT, kT, mask, scores);

  softmax_rows_kernel<<<dim3(kN, kB), NT, 0, stream>>>(scores, attnM, vis);

  gemm_av_kernel<<<dim3(kN / BN, kC / BM, kB), NT, 0, stream>>>(
      vM, attnM, x, gamma, out);

  vis_norm_kernel<<<kB, NT, 0, stream>>>(vis, visOut);
}
